// CrossAttention_65566970740946
// MI455X (gfx1250) — compile-verified, matrix operand order checked by reading, not run on hardware
//
#include <hip/hip_runtime.h>
#include <hip/hip_bf16.h>

typedef __attribute__((ext_vector_type(16))) __bf16 v16bf;
typedef __attribute__((ext_vector_type(8)))  float  v8f;

#define TPB 256

static __device__ __forceinline__ unsigned short f2bf(float f) {
  unsigned u = __float_as_uint(f);
  unsigned r = u + 0x7FFFu + ((u >> 16) & 1u);
  return (unsigned short)(r >> 16);
}

static __device__ __forceinline__ v8f wmma_bf16(v16bf a, v16bf b, v8f c) {
  // D = A(16x32) * B(32x16) + C, f32 accumulate
  return __builtin_amdgcn_wmma_f32_16x16x32_bf16(false, a, false, b, (short)0, c, false, false);
}

// ASYNCcnt-tracked 16B copy global -> LDS (GLOBAL_LOAD_ASYNC_TO_LDS_B128, GVS mode).
// lds_off: LDS byte address (VGPR), base: SGPR-pair global base, voff: byte offset (VGPR).
static __device__ __forceinline__ void async_cp_b128(unsigned lds_off, const void* base,
                                                     unsigned voff) {
  asm volatile("global_load_async_to_lds_b128 %0, %1, %2"
               :: "v"(lds_off), "v"(voff), "s"(base)
               : "memory");
}
static __device__ __forceinline__ void async_wait0() {
  asm volatile("s_wait_asynccnt 0" ::: "memory");
}

// Load one 16x32 bf16 fragment from an LDS tile stored [row][k] with leading dim `ld`.
// Per CDNA5 ISA: lanes 0-15 hold row (base+lane), elems = K {0..7,16..23}; lanes 16-31
// hold the same rows with K {8..15,24..31}.  Two 16B LDS reads per fragment.
static __device__ __forceinline__ v16bf frag_ld(const unsigned short* base, int mrow,
                                                int koff, int ld, int lane) {
  const unsigned short* p = base + (size_t)(mrow + (lane & 15)) * ld + koff + ((lane >> 4) << 3);
  union { uint4 u[2]; v16bf v; } x;
  x.u[0] = *(const uint4*)p;
  x.u[1] = *(const uint4*)(p + 16);
  return x.v;
}

// ---------------------------------------------------------------------------
// Tiled WMMA GEMM: C[M,N] = A[M,K](bf16) * B[K,N](f32 weights, converted on load)
// Block tile 128x128, 8 waves (4M x 2N), each wave 32x64 = 2x4 wmma tiles, K-step 32.
// A tile copied via global_load_async_to_lds_b128 (ASYNCcnt); B converted in-register.
// MODE 0: out = bf16((acc + bias[col]) * scale), row remapped b*SKVtot+row_off+s
// MODE 1: out(f32) = aux + tanh(gate)*(acc + bias[col])
// MODE 2: out = bf16(gelu_exact(acc))
// MODE 3: out(f32) = aux + tanh(gate)*acc
// ---------------------------------------------------------------------------
template <int MODE>
__global__ __launch_bounds__(TPB)
void gemm_wmma(const unsigned short* __restrict__ A, const float* __restrict__ Bw,
               const float* __restrict__ bias, const float* __restrict__ aux,
               const float* __restrict__ gate, void* __restrict__ outp,
               int M, int N, int K, int Sin, int SKVtot, int row_off, float scale) {
  __shared__ __align__(16) unsigned short As[128 * 40]; // [m][k], ld=40
  __shared__ __align__(16) unsigned short Bs[128 * 40]; // [n][k], ld=40 (transposed)

  const int tid = threadIdx.x, lane = tid & 31, w = tid >> 5;
  const int m0 = blockIdx.y * 128, n0 = blockIdx.x * 128;
  const int wm = (w >> 1) * 32, wn = (w & 1) * 64;

  v8f acc[2][4] = {};

  // A: 2 threads/row, 16 elems (32B) each -> two async b128 copies per thread
  const int ar = tid >> 1, akb = (tid & 1) * 16;
  const unsigned a_lds0 = (unsigned)(uintptr_t)&As[ar * 40 + akb];
  const unsigned a_gbase = ((unsigned)(m0 + ar) * (unsigned)K + (unsigned)akb) * 2u;
  // B: 8 threads/row (k), 16 cols each
  const int bk = tid >> 3, bn = (tid & 7) * 16;

  for (int k0 = 0; k0 < K; k0 += 32) {
    // async copy A tile (LDS writes are safe: previous iter consumers done at loop-end sync)
    unsigned avoff = a_gbase + (unsigned)k0 * 2u;
    async_cp_b128(a_lds0,       A, avoff);
    async_cp_b128(a_lds0 + 16u, A, avoff + 16u);

    // stage B through registers with f32->bf16 conversion, scatter transposed
    const float* gb = Bw + (size_t)(k0 + bk) * N + n0 + bn;
    float4 f0 = *(const float4*)gb;
    float4 f1 = *(const float4*)(gb + 4);
    float4 f2 = *(const float4*)(gb + 8);
    float4 f3 = *(const float4*)(gb + 12);
    unsigned short qb[16] = { f2bf(f0.x), f2bf(f0.y), f2bf(f0.z), f2bf(f0.w),
                              f2bf(f1.x), f2bf(f1.y), f2bf(f1.z), f2bf(f1.w),
                              f2bf(f2.x), f2bf(f2.y), f2bf(f2.z), f2bf(f2.w),
                              f2bf(f3.x), f2bf(f3.y), f2bf(f3.z), f2bf(f3.w) };
#pragma unroll
    for (int j = 0; j < 16; ++j) Bs[(bn + j) * 40 + bk] = qb[j];

    async_wait0();      // our A-tile copies have landed in LDS
    __syncthreads();    // all waves' A copies + B stores visible

    v16bf a_lo = frag_ld(As, wm,      0, 40, lane);
    v16bf a_hi = frag_ld(As, wm + 16, 0, 40, lane);
#pragma unroll
    for (int ni = 0; ni < 4; ++ni) {
      v16bf bf_ = frag_ld(Bs, wn + ni * 16, 0, 40, lane);
      acc[0][ni] = wmma_bf16(a_lo, bf_, acc[0][ni]);
      acc[1][ni] = wmma_bf16(a_hi, bf_, acc[1][ni]);
    }
    __syncthreads();
  }

  const int hi8 = (lane >> 4) << 3, nc = lane & 15;
  float gt = 0.f;
  if (MODE == 1 || MODE == 3) gt = tanhf(gate[0]);

#pragma unroll
  for (int mi = 0; mi < 2; ++mi)
#pragma unroll
    for (int ni = 0; ni < 4; ++ni)
#pragma unroll
      for (int r = 0; r < 8; ++r) {
        int row = m0 + wm + mi * 16 + r + hi8;
        int col = n0 + wn + ni * 16 + nc;
        float v = acc[mi][ni][r];
        if (MODE == 0) {
          v = (v + bias[col]) * scale;
          int bb = row / Sin, ss = row - bb * Sin;
          size_t orow = (size_t)bb * SKVtot + row_off + ss;
          ((unsigned short*)outp)[orow * N + col] = f2bf(v);
        } else if (MODE == 1) {
          size_t idx = (size_t)row * N + col;
          ((float*)outp)[idx] = aux[idx] + gt * (v + bias[col]);
        } else if (MODE == 2) {
          float g = 0.5f * v * (1.f + erff(v * 0.70710678118f));
          ((unsigned short*)outp)[(size_t)row * N + col] = f2bf(g);
        } else {
          size_t idx = (size_t)row * N + col;
          ((float*)outp)[idx] = aux[idx] + gt * v;
        }
      }
}

// ---------------------------------------------------------------------------
// Flash attention: one block per (b, h, 32 query rows). HD=128, KV chunks of 64.
// K chunk copied via async-to-LDS; V staged through registers (needs transpose).
// ---------------------------------------------------------------------------
__global__ __launch_bounds__(TPB)
void attn_flash(const unsigned short* __restrict__ Qg, const unsigned short* __restrict__ Kg,
                const unsigned short* __restrict__ Vg, const float* __restrict__ qmask,
                const float* __restrict__ kvmask, unsigned short* __restrict__ ctx,
                int SQ, int SKV, int D) {
  __shared__ __align__(16) unsigned short Qs[32 * 136]; // [q][hd]
  __shared__ __align__(16) unsigned short Ks[64 * 136]; // [kv][hd] (== [n][k] for QK^T)
  __shared__ __align__(16) unsigned short Vt[128 * 72]; // [hd][kv] (== [n][k] for P·V)
  __shared__ __align__(16) float          Sc[32 * 68];  // f32 scores
  __shared__ __align__(16) unsigned short Ps[32 * 72];  // bf16 probs
  __shared__ float mrow[32], lrow[32], rrow[32];

  const int tid = threadIdx.x, lane = tid & 31, w = tid >> 5;
  const int qb = blockIdx.x, h = blockIdx.y, b = blockIdx.z;
  const int q0 = qb * 32;
  const size_t qbase = ((size_t)b * SQ + q0) * D + h * 128;
  const size_t kvb   = (size_t)b * SKV * D + h * 128;

  { // load Q tile: 32x128, 16 elems/thread
    int r = tid >> 3, kb = (tid & 7) * 16;
    const unsigned short* g = Qg + qbase + (size_t)r * D + kb;
    *(uint4*)&Qs[r * 136 + kb]     = *(const uint4*)g;
    *(uint4*)&Qs[r * 136 + kb + 8] = *(const uint4*)(g + 8);
  }
  if (tid < 32) { mrow[tid] = -3.0e38f; lrow[tid] = 0.f; }

  const int smt = (w >> 2) * 16, snt = (w & 3) * 16; // scores: 1 tile / wave
  const int omt = smt, onb = (w & 3) * 32;           // output: 2 tiles / wave
  const int hi8 = (lane >> 4) << 3, nc = lane & 15;
  v8f oacc[2] = {};

  // per-thread K/V chunk mapping: 4 threads/row, 32 elems (64B) each
  const int cr = tid >> 2, ckb = (tid & 3) * 32;
  const unsigned k_lds0 = (unsigned)(uintptr_t)&Ks[cr * 136 + ckb];

  for (int kv0 = 0; kv0 < SKV; kv0 += 64) {
    __syncthreads(); // previous-iteration consumers done before LDS overwrite
    { // K chunk 64x128 via async copies (4 x b128 per thread)
      unsigned kvoff = (unsigned)((kvb + (size_t)(kv0 + cr) * D + ckb) * 2);
#pragma unroll
      for (int j = 0; j < 4; ++j)
        async_cp_b128(k_lds0 + j * 16u, Kg, kvoff + j * 16u);
    }
    { // V chunk, transposed into Vt[hd][kv]
      const unsigned short* g = Vg + kvb + (size_t)(kv0 + cr) * D + ckb;
#pragma unroll
      for (int j = 0; j < 4; ++j) {
        union { uint4 u; unsigned short s[8]; } x;
        x.u = *(const uint4*)(g + j * 8);
#pragma unroll
        for (int e = 0; e < 8; ++e) Vt[(ckb + j * 8 + e) * 72 + cr] = x.s[e];
      }
    }
    async_wait0();
    __syncthreads();
    { // scores: 16x16 tile per wave, K = 128
      v8f s = {};
#pragma unroll
      for (int ks = 0; ks < 4; ++ks) {
        v16bf a  = frag_ld(Qs, smt, ks * 32, 136, lane);
        v16bf bb = frag_ld(Ks, snt, ks * 32, 136, lane);
        s = wmma_bf16(a, bb, s);
      }
#pragma unroll
      for (int r = 0; r < 8; ++r) Sc[(smt + r + hi8) * 68 + snt + nc] = s[r];
    }
    __syncthreads();
    { // online softmax: 8 threads per query row
      int row = tid >> 3, j = tid & 7;
      float qm = qmask[(size_t)b * SQ + q0 + row];
      float vals[8], pm = -3.0e38f;
#pragma unroll
      for (int c = 0; c < 8; ++c) {
        int col = j + c * 8;
        float s = Sc[row * 68 + col];
        float mk = kvmask[(size_t)b * SKV + kv0 + col] * qm;
        if (mk == 0.f) s = -3.0e38f;
        vals[c] = s;
        pm = fmaxf(pm, s);
      }
#pragma unroll
      for (int o = 4; o > 0; o >>= 1) pm = fmaxf(pm, __shfl_xor(pm, o, 8));
      float mprev = mrow[row];
      float mnew  = fmaxf(mprev, pm);
      float resc  = __expf(mprev - mnew);
      float psum  = 0.f;
#pragma unroll
      for (int c = 0; c < 8; ++c) {
        float p = __expf(vals[c] - mnew);
        psum += p;
        Ps[row * 72 + j + c * 8] = f2bf(p);
      }
#pragma unroll
      for (int o = 4; o > 0; o >>= 1) psum += __shfl_xor(psum, o, 8);
      if (j == 0) { mrow[row] = mnew; lrow[row] = lrow[row] * resc + psum; rrow[row] = resc; }
    }
    __syncthreads();
    // rescale running output, then accumulate P(32x64) · V(64x128)
#pragma unroll
    for (int t = 0; t < 2; ++t) {
#pragma unroll
      for (int r = 0; r < 8; ++r) oacc[t][r] *= rrow[omt + r + hi8];
      int nt = onb + t * 16;
#pragma unroll
      for (int ks = 0; ks < 2; ++ks) {
        v16bf a  = frag_ld(Ps, omt, ks * 32, 72, lane);
        v16bf bb = frag_ld(Vt, nt,  ks * 32, 72, lane);
        oacc[t] = wmma_bf16(a, bb, oacc[t]);
      }
    }
  }
  __syncthreads();
#pragma unroll
  for (int t = 0; t < 2; ++t) {
    int nt = onb + t * 16;
#pragma unroll
    for (int r = 0; r < 8; ++r) {
      int row = omt + r + hi8;
      float l = fmaxf(lrow[row], 1e-20f);
      float v = oacc[t][r] / l;
      ctx[((size_t)b * SQ + q0 + row) * D + h * 128 + nt + nc] = f2bf(v);
    }
  }
}

// ---------------------------------------------------------------------------
// Norm + helper kernels
// ---------------------------------------------------------------------------
__global__ __launch_bounds__(TPB)
void rmsnorm_to_bf16(const float* __restrict__ x, const float* __restrict__ wgt,
                     unsigned short* __restrict__ out, int D) {
  const int row = blockIdx.x;
  const float* xr = x + (size_t)row * D;
  float ss = 0.f;
  for (int i = threadIdx.x; i < D; i += TPB) { float v = xr[i]; ss += v * v; }
  __shared__ float red[9];
  int lane = threadIdx.x & 31, w = threadIdx.x >> 5;
#pragma unroll
  for (int o = 16; o > 0; o >>= 1) ss += __shfl_xor(ss, o, 32);
  if (lane == 0) red[w] = ss;
  __syncthreads();
  if (threadIdx.x == 0) { float t = 0.f; for (int i = 0; i < 8; ++i) t += red[i]; red[8] = t; }
  __syncthreads();
  float inv = rsqrtf(red[8] / (float)D + 1e-6f);
  for (int i = threadIdx.x; i < D; i += TPB)
    out[(size_t)row * D + i] = f2bf(xr[i] * inv * wgt[i]);
}

__global__ __launch_bounds__(TPB)
void layernorm_to_bf16(const float* __restrict__ x, const float* __restrict__ g,
                       const float* __restrict__ bta, unsigned short* __restrict__ out, int D) {
  const int row = blockIdx.x;
  const float* xr = x + (size_t)row * D;
  float s1 = 0.f, s2 = 0.f;
  for (int i = threadIdx.x; i < D; i += TPB) { float v = xr[i]; s1 += v; s2 += v * v; }
  __shared__ float r1[9], r2[9];
  int lane = threadIdx.x & 31, w = threadIdx.x >> 5;
#pragma unroll
  for (int o = 16; o > 0; o >>= 1) { s1 += __shfl_xor(s1, o, 32); s2 += __shfl_xor(s2, o, 32); }
  if (lane == 0) { r1[w] = s1; r2[w] = s2; }
  __syncthreads();
  if (threadIdx.x == 0) {
    float t1 = 0.f, t2 = 0.f;
    for (int i = 0; i < 8; ++i) { t1 += r1[i]; t2 += r2[i]; }
    r1[8] = t1; r2[8] = t2;
  }
  __syncthreads();
  float mu  = r1[8] / (float)D;
  float var = r2[8] / (float)D - mu * mu;
  float inv = rsqrtf(var + 1e-5f);
  for (int i = threadIdx.x; i < D; i += TPB)
    out[(size_t)row * D + i] = f2bf((xr[i] - mu) * inv * g[i] + bta[i]);
}

__global__ void f32_to_bf16_k(const float* __restrict__ in, unsigned short* __restrict__ out,
                              size_t n) {
  size_t i = (size_t)blockIdx.x * blockDim.x + threadIdx.x;
  size_t stride = (size_t)gridDim.x * blockDim.x;
  for (; i < n; i += stride) out[i] = f2bf(in[i]);
}

__global__ void concat_mask_k(const float* __restrict__ pm, const float* __restrict__ sm,
                              const float* __restrict__ mm, float* __restrict__ out,
                              int B_, int SP_, int SS_, int SM_) {
  int skv = SP_ + SS_ + SM_;
  int i = blockIdx.x * blockDim.x + threadIdx.x;
  if (i >= B_ * skv) return;
  int b = i / skv, j = i - b * skv;
  float v;
  if (j < SP_)            v = pm[b * SP_ + j];
  else if (j < SP_ + SS_) v = sm[b * SS_ + (j - SP_)];
  else                    v = mm[b * SM_ + (j - SP_ - SS_)];
  out[i] = v;
}

// ---------------------------------------------------------------------------
extern "C" void kernel_launch(void* const* d_in, const int* in_sizes, int n_in,
                              void* d_out, int out_size, void* d_ws, size_t ws_size,
                              hipStream_t stream) {
  (void)in_sizes; (void)n_in; (void)out_size; (void)ws_size;
  const int Bc = 2, SQ = 1024, Dm = 2048, Hh = 16;
  const int SP = 1024, SSd = 1024, SMd = 512, SKV = 2560;
  const int DP = 1280, DSd = 1024, DMd = 768, INNER = 8192;

  const float* q_in   = (const float*)d_in[0];
  const float* pkv_in = (const float*)d_in[1];
  const float* skv_in = (const float*)d_in[2];
  const float* mkv_in = (const float*)d_in[3];
  const float* qmask  = (const float*)d_in[4];
  const float* pmask  = (const float*)d_in[5];
  const float* smask  = (const float*)d_in[6];
  const float* mmask  = (const float*)d_in[7];
  const float* rmsw   = (const float*)d_in[8];
  const float* Wq  = (const float*)d_in[9];  const float* bq  = (const float*)d_in[10];
  const float* Wkp = (const float*)d_in[11]; const float* bkp = (const float*)d_in[12];
  const float* Wvp = (const float*)d_in[13]; const float* bvp = (const float*)d_in[14];
  const float* Wks = (const float*)d_in[15]; const float* bks = (const float*)d_in[16];
  const float* Wvs = (const float*)d_in[17]; const float* bvs = (const float*)d_in[18];
  const float* Wkm = (const float*)d_in[19]; const float* bkm = (const float*)d_in[20];
  const float* Wvm = (const float*)d_in[21]; const float* bvm = (const float*)d_in[22];
  const float* Wo  = (const float*)d_in[23]; const float* bo  = (const float*)d_in[24];
  const float* lng = (const float*)d_in[25]; const float* lnb = (const float*)d_in[26];
  const float* W1  = (const float*)d_in[27]; const float* W2  = (const float*)d_in[28];
  const float* g_attn = (const float*)d_in[29];
  const float* g_ffw  = (const float*)d_in[30];

  char* ws = (char*)d_ws;
  size_t off = 0;
  auto alloc = [&](size_t bytes) -> void* {
    void* p = ws + off;
    off += (bytes + 255) & ~(size_t)255;
    return p;
  };
  unsigned short* xn16  = (unsigned short*)alloc((size_t)Bc * SQ  * Dm    * 2);
  unsigned short* p16   = (unsigned short*)alloc((size_t)Bc * SP  * DP    * 2);
  unsigned short* s16   = (unsigned short*)alloc((size_t)Bc * SSd * DSd   * 2);
  unsigned short* m16   = (unsigned short*)alloc((size_t)Bc * SMd * DMd   * 2);
  unsigned short* q16   = (unsigned short*)alloc((size_t)Bc * SQ  * Dm    * 2);
  unsigned short* k16   = (unsigned short*)alloc((size_t)Bc * SKV * Dm    * 2);
  unsigned short* v16   = (unsigned short*)alloc((size_t)Bc * SKV * Dm    * 2);
  unsigned short* ctx16 = (unsigned short*)alloc((size_t)Bc * SQ  * Dm    * 2);
  unsigned short* hn16  = (unsigned short*)alloc((size_t)Bc * SQ  * Dm    * 2);
  unsigned short* ge16  = (unsigned short*)alloc((size_t)Bc * SQ  * INNER * 2);
  float*          h32   = (float*)alloc((size_t)Bc * SQ * Dm * 4);
  float*          kvm   = (float*)alloc((size_t)Bc * SKV * 4);

  // down-convert KV inputs, build concat mask, RMS-norm the query stream
  f32_to_bf16_k<<<512, TPB, 0, stream>>>(pkv_in, p16, (size_t)Bc * SP  * DP);
  f32_to_bf16_k<<<512, TPB, 0, stream>>>(skv_in, s16, (size_t)Bc * SSd * DSd);
  f32_to_bf16_k<<<512, TPB, 0, stream>>>(mkv_in, m16, (size_t)Bc * SMd * DMd);
  concat_mask_k<<<(Bc * SKV + TPB - 1) / TPB, TPB, 0, stream>>>(pmask, smask, mmask, kvm,
                                                                Bc, SP, SSd, SMd);
  rmsnorm_to_bf16<<<Bc * SQ, TPB, 0, stream>>>(q_in, rmsw, xn16, Dm);

  // Q/K/V projections (scale H^-0.5 = 0.25 folded into Q)
  dim3 gq(Dm / 128, (Bc * SQ) / 128);
  gemm_wmma<0><<<gq, TPB, 0, stream>>>(xn16, Wq, bq, nullptr, nullptr, q16,
                                       Bc * SQ, Dm, Dm, Bc * SQ, Bc * SQ, 0, 0.25f);
  dim3 gp(Dm / 128, (Bc * SP) / 128);
  gemm_wmma<0><<<gp, TPB, 0, stream>>>(p16, Wkp, bkp, nullptr, nullptr, k16,
                                       Bc * SP, Dm, DP, SP, SKV, 0, 1.f);
  gemm_wmma<0><<<gp, TPB, 0, stream>>>(p16, Wvp, bvp, nullptr, nullptr, v16,
                                       Bc * SP, Dm, DP, SP, SKV, 0, 1.f);
  dim3 gs(Dm / 128, (Bc * SSd) / 128);
  gemm_wmma<0><<<gs, TPB, 0, stream>>>(s16, Wks, bks, nullptr, nullptr, k16,
                                       Bc * SSd, Dm, DSd, SSd, SKV, SP, 1.f);
  gemm_wmma<0><<<gs, TPB, 0, stream>>>(s16, Wvs, bvs, nullptr, nullptr, v16,
                                       Bc * SSd, Dm, DSd, SSd, SKV, SP, 1.f);
  dim3 gm(Dm / 128, (Bc * SMd) / 128);
  gemm_wmma<0><<<gm, TPB, 0, stream>>>(m16, Wkm, bkm, nullptr, nullptr, k16,
                                       Bc * SMd, Dm, DMd, SMd, SKV, SP + SSd, 1.f);
  gemm_wmma<0><<<gm, TPB, 0, stream>>>(m16, Wvm, bvm, nullptr, nullptr, v16,
                                       Bc * SMd, Dm, DMd, SMd, SKV, SP + SSd, 1.f);

  // flash attention
  dim3 ga(SQ / 32, Hh, Bc);
  attn_flash<<<ga, TPB, 0, stream>>>(q16, k16, v16, qmask, kvm, ctx16, SQ, SKV, Dm);

  // output projection + gated residual -> h (f32)
  gemm_wmma<1><<<gq, TPB, 0, stream>>>(ctx16, Wo, bo, q_in, g_attn, h32,
                                       Bc * SQ, Dm, Dm, Bc * SQ, Bc * SQ, 0, 1.f);
  // layernorm -> bf16
  layernorm_to_bf16<<<Bc * SQ, TPB, 0, stream>>>(h32, lng, lnb, hn16, Dm);
  // FFW up + exact GELU
  dim3 g1(INNER / 128, (Bc * SQ) / 128);
  gemm_wmma<2><<<g1, TPB, 0, stream>>>(hn16, W1, nullptr, nullptr, nullptr, ge16,
                                       Bc * SQ, INNER, Dm, Bc * SQ, Bc * SQ, 0, 1.f);
  // FFW down + gated residual -> final output (f32)
  gemm_wmma<3><<<gq, TPB, 0, stream>>>(ge16, W2, nullptr, h32, g_ffw, (float*)d_out,
                                       Bc * SQ, Dm, INNER, Bc * SQ, Bc * SQ, 0, 1.f);
}
